// MultiHeadAttention_30932354466047
// MI455X (gfx1250) — matrix operand order repaired and checked by reading, not compiled, not run on hardware
//
#include <hip/hip_runtime.h>
#include <hip/hip_bf16.h>

// ---------------------------------------------------------------------------
// MI455X (gfx1250) multi-head attention, bf16 WMMA (fp32 accum) end-to-end.
// D_MODEL=1024, H=16, DK=64, B=4, N=2048, causal + padding mask.
// ---------------------------------------------------------------------------

typedef __attribute__((ext_vector_type(16))) __bf16          v16bf;
typedef __attribute__((ext_vector_type(8)))  float           v8f;
typedef __attribute__((ext_vector_type(8)))  float           f32x8;
typedef __attribute__((ext_vector_type(8)))  unsigned short  u16x8;
typedef __attribute__((ext_vector_type(16))) unsigned short  u16x16;

union BF16Frag {
    u16x16 u;
    v16bf  v;
    u16x8  h[2];
};

static __device__ __forceinline__ unsigned short f2bf(float f) {
    unsigned int u = __builtin_bit_cast(unsigned int, f);
    u += 0x7FFFu + ((u >> 16) & 1u);        // round-to-nearest-even
    return (unsigned short)(u >> 16);
}

static __device__ __forceinline__ v8f wmma_bf16(v16bf a, v16bf b, v8f c) {
    // v_wmma_f32_16x16x32_bf16: D = A(16x32) * B(32x16) + C(16x16)
    return __builtin_amdgcn_wmma_f32_16x16x32_bf16(
        /*neg_a=*/false, a, /*neg_b=*/false, b,
        /*c_mod=*/(short)0, c, /*reuse_a=*/false, /*reuse_b=*/false);
}

// ---------------------------------------------------------------------------
// Kernel 1: transpose + convert a 1024x1024 f32 weight (k-major) into
// bf16 Wt[n][k] (k contiguous) so GEMM B-fragments are two contiguous 16B runs.
// ---------------------------------------------------------------------------
__global__ __launch_bounds__(256) void wtrans_kernel(const float* __restrict__ W,
                                                     unsigned short* __restrict__ Wt) {
    __shared__ float t[32][33];
    const int n = blockIdx.x * 32 + threadIdx.x;   // column of W
    const int k = blockIdx.y * 32 + threadIdx.y;   // row of W
#pragma unroll
    for (int i = 0; i < 32; i += 8)
        t[threadIdx.y + i][threadIdx.x] = W[(size_t)(k + i) * 1024 + n];
    __syncthreads();
    const int k2 = blockIdx.y * 32 + threadIdx.x;
    const int n2 = blockIdx.x * 32 + threadIdx.y;
#pragma unroll
    for (int i = 0; i < 32; i += 8)
        Wt[(size_t)(n2 + i) * 1024 + k2] = f2bf(t[threadIdx.x][threadIdx.y + i]);
}

// ---------------------------------------------------------------------------
// Kernel 2: WMMA GEMM  C[M,N] = A[M,K] * Bt[N,K]^T  (Bt is n-major bf16).
// One wave computes a 64x64 tile (4x4 WMMA accum tiles, A reuse x4,
// B reuse x4), K stepped by 32. B fragments consumed one at a time to
// bound register pressure.
//   AF32 : A operand is f32 (converted to bf16 inline) vs bf16
//   MODE : 0 = bf16 out, token-major [M][N]
//          1 = bf16 out scattered as Vt[b][h][d][tok]   (for the V projection)
//          2 = f32  out, token-major [M][N]             (final output)
// ---------------------------------------------------------------------------
template <int AF32, int MODE>
__global__ __launch_bounds__(256) void gemm_wmma_kernel(const void* __restrict__ Ap,
                                                        const unsigned short* __restrict__ Bt,
                                                        void* __restrict__ Cp,
                                                        int M, int N, int K) {
    const int wave = threadIdx.x >> 5;
    const int lane = threadIdx.x & 31;
    const int half = lane >> 4;          // which 16-lane half
    const int r    = lane & 15;
    const int kb    = half * 8;          // A-fragment K base per half
    const int khalf = half * 16;         // unused; B fragments use kb like A

    const int wid    = blockIdx.x * 8 + wave;
    const int tilesN = N / 64;
    const int m0 = (wid / tilesN) * 64;
    const int n0 = (wid % tilesN) * 64;

    v8f acc[4][4];
#pragma unroll
    for (int im = 0; im < 4; ++im)
#pragma unroll
        for (int in_ = 0; in_ < 4; ++in_)
#pragma unroll
            for (int e = 0; e < 8; ++e) acc[im][in_][e] = 0.0f;

    for (int k0 = 0; k0 < K; k0 += 32) {
        BF16Frag a[4];
        if (AF32) {
            const float* Af = (const float*)Ap;
#pragma unroll
            for (int im = 0; im < 4; ++im) {
                const float* ar = Af + (size_t)(m0 + im * 16 + r) * K + k0;
                f32x8 x0 = *(const f32x8*)(ar + kb);
                f32x8 x1 = *(const f32x8*)(ar + kb + 16);
#pragma unroll
                for (int e = 0; e < 8; ++e) {
                    a[im].u[e]     = f2bf(x0[e]);
                    a[im].u[8 + e] = f2bf(x1[e]);
                }
            }
        } else {
            const unsigned short* Ab = (const unsigned short*)Ap;
#pragma unroll
            for (int im = 0; im < 4; ++im) {
                const unsigned short* ar = Ab + (size_t)(m0 + im * 16 + r) * K + k0;
                a[im].h[0] = *(const u16x8*)(ar + kb);
                a[im].h[1] = *(const u16x8*)(ar + kb + 16);
            }
        }
#pragma unroll
        for (int in_ = 0; in_ < 4; ++in_) {
            BF16Frag bfr;
            bfr.h[0] = *(const u16x8*)(Bt + (size_t)(n0 + in_ * 16 + r) * K + k0 + kb);
            bfr.h[1] = *(const u16x8*)(Bt + (size_t)(n0 + in_ * 16 + r) * K + k0 + kb + 16);
#pragma unroll
            for (int im = 0; im < 4; ++im)
                acc[im][in_] = wmma_bf16(a[im].v, bfr.v, acc[im][in_]);
        }
    }

#pragma unroll
    for (int im = 0; im < 4; ++im)
#pragma unroll
        for (int in_ = 0; in_ < 4; ++in_)
#pragma unroll
            for (int v = 0; v < 8; ++v) {
                const int gm = m0 + im * 16 + v + half * 8;   // token row
                const int gn = n0 + in_ * 16 + r;             // feature col
                const float val = acc[im][in_][v];
                if (MODE == 0) {
                    ((unsigned short*)Cp)[(size_t)gm * N + gn] = f2bf(val);
                } else if (MODE == 1) {
                    const int bb = gm >> 11, tok = gm & 2047;
                    const int hh = gn >> 6,  dd  = gn & 63;
                    ((unsigned short*)Cp)[(((size_t)(bb * 16 + hh) * 64 + dd) << 11) + tok] =
                        f2bf(val);
                } else {
                    ((float*)Cp)[(size_t)gm * N + gn] = val;
                }
            }
}

// ---------------------------------------------------------------------------
// Kernel 3: flash attention. One wave per (b, h, 32-query tile): each K/V
// fragment loaded once feeds two score / two AV WMMAs. Online base-2
// softmax; P restaged via LDS (C->A layout). The softmax denominator is
// accumulated on the MATRIX unit: a 5th accumulator tile per query sub-tile
// is fed by wmma(P, ones) — its recurrence (rescale by corr, add row sum)
// is identical to O's, and every lane of the C tile holds its row's sum,
// eliminating the 4-step ds_bpermute sum butterfly entirely. Only the
// row-max reduction uses shuffles. Causal split: bulk chunks carry no
// causal compares; only the single diagonal chunk does per-element masking.
// ---------------------------------------------------------------------------
__global__ __launch_bounds__(256) void attn_kernel(const unsigned short* __restrict__ Qb,
                                                   const unsigned short* __restrict__ Kb,
                                                   const unsigned short* __restrict__ Vt,
                                                   const int* __restrict__ maskp,
                                                   unsigned short* __restrict__ Ob) {
    __shared__ unsigned short ldsP[8][2][16 * 32];   // per-wave two 16x32 bf16 P tiles

    const int wave = threadIdx.x >> 5;
    const int lane = threadIdx.x & 31;
    const int half = lane >> 4;
    const int r    = lane & 15;
    const int kb   = half * 8;

    const int wid    = blockIdx.x * 8 + wave;
    const int qtiles = 2048 / 32;                  // 64
    const int b   = wid / (16 * qtiles);
    const int rem = wid % (16 * qtiles);
    const int h   = rem / qtiles;
    const int q0  = (rem % qtiles) * 32;           // first of 32 queries

    // Q A-fragments: 2 query sub-tiles x 2 d-chunks, kept in registers.
    BF16Frag qf[2][2];
#pragma unroll
    for (int im = 0; im < 2; ++im) {
        const unsigned short* qrow =
            Qb + (size_t)(b * 2048 + q0 + im * 16 + r) * 1024 + h * 64;
#pragma unroll
        for (int c = 0; c < 2; ++c) {
            qf[im][c].h[0] = *(const u16x8*)(qrow + c * 32 + kb);
            qf[im][c].h[1] = *(const u16x8*)(qrow + c * 32 + kb + 16);
        }
    }

    // All-ones B fragment: feeds the row-sum (denominator) accumulator.
    BF16Frag ones;
#pragma unroll
    for (int e = 0; e < 16; ++e) ones.u[e] = 0x3F80;   // bf16 1.0

    v8f O[2][5];                                   // [im][0..3]=output, [4]=row sums
#pragma unroll
    for (int im = 0; im < 2; ++im)
#pragma unroll
        for (int t = 0; t < 5; ++t)
#pragma unroll
            for (int e = 0; e < 8; ++e) O[im][t][e] = 0.0f;

    float mrow[2][8];
#pragma unroll
    for (int im = 0; im < 2; ++im)
#pragma unroll
        for (int v = 0; v < 8; ++v) mrow[im][v] = -1e30f;

    const float sc = 0.125f * 1.44269504088896340736f;  // 1/sqrt(64) * log2(e)
    const int jend = q0 + 32;                            // causal bound (exclusive)
    // q0 is a multiple of 32: chunks j0 < q0 are fully unmasked, the single
    // chunk j0 == q0 is the diagonal one.
    const int jbulk = q0;

    const size_t kbase = (size_t)(b * 2048) * 1024 + h * 64 + r * 1024;
    const size_t vbase = (size_t)((b * 16 + h) * 64 + r) * 2048 + half * 16;

    for (int j0 = 0; j0 < jend; j0 += 32) {
        const bool diag = (j0 >= jbulk);

        // ---- scores: 2 key tiles x 2 query tiles; K frags loaded once ----
        v8f S[2][2];                                 // [im][jt]
#pragma unroll
        for (int jt = 0; jt < 2; ++jt) {
            const unsigned short* krow = Kb + kbase + (size_t)(j0 + jt * 16) * 1024;
            BF16Frag kf[2];
#pragma unroll
            for (int c = 0; c < 2; ++c)
            {
                kf[c].h[0] = *(const u16x8*)(krow + c * 32 + kb);
                kf[c].h[1] = *(const u16x8*)(krow + c * 32 + kb + 16);
            }
#pragma unroll
            for (int im = 0; im < 2; ++im) {
                v8f s;
#pragma unroll
                for (int e = 0; e < 8; ++e) s[e] = 0.0f;
                s = wmma_bf16(qf[im][0].v, kf[0].v, s);
                s = wmma_bf16(qf[im][1].v, kf[1].v, s);
                S[im][jt] = s;
            }
        }

        // ---- scale + masks (scores kept in log2 domain) ----
        const float bias0 = (maskp[b * 2048 + j0 + r]      == 0) ? -1e30f : 0.0f;
        const float bias1 = (maskp[b * 2048 + j0 + 16 + r] == 0) ? -1e30f : 0.0f;
        if (diag) {
            const int jA = j0 + r, jB = j0 + 16 + r;
#pragma unroll
            for (int im = 0; im < 2; ++im)
#pragma unroll
                for (int v = 0; v < 8; ++v) {
                    const int q = q0 + im * 16 + v + half * 8;
                    float s0 = __builtin_fmaf(S[im][0][v], sc, bias0);
                    float s1 = __builtin_fmaf(S[im][1][v], sc, bias1);
                    if (jA > q) s0 = -1e30f;
                    if (jB > q) s1 = -1e30f;
                    S[im][0][v] = s0;
                    S[im][1][v] = s1;
                }
        } else {
#pragma unroll
            for (int im = 0; im < 2; ++im)
#pragma unroll
                for (int v = 0; v < 8; ++v) {
                    S[im][0][v] = __builtin_fmaf(S[im][0][v], sc, bias0);
                    S[im][1][v] = __builtin_fmaf(S[im][1][v], sc, bias1);
                }
        }

        // ---- online softmax: only the row-max needs a shuffle butterfly ----
#pragma unroll
        for (int im = 0; im < 2; ++im)
#pragma unroll
            for (int v = 0; v < 8; ++v) {
                float rm = fmaxf(S[im][0][v], S[im][1][v]);
#pragma unroll
                for (int o = 1; o < 16; o <<= 1) rm = fmaxf(rm, __shfl_xor(rm, o, 32));
                const float mn   = fmaxf(mrow[im][v], rm);
                const float corr = exp2f(mrow[im][v] - mn);
                mrow[im][v] = mn;
                const float p0 = exp2f(S[im][0][v] - mn);
                const float p1 = exp2f(S[im][1][v] - mn);
#pragma unroll
                for (int t = 0; t < 5; ++t) O[im][t][v] *= corr;
                // stage P (C layout) into LDS as bf16, row m = v + half*8
                const int m = v + half * 8;
                ldsP[wave][im][m * 32 + r]      = f2bf(p0);
                ldsP[wave][im][m * 32 + 16 + r] = f2bf(p1);
            }

        // wave-synchronous LDS exchange: drain DS ops before re-reading
        asm volatile("s_wait_dscnt 0" ::: "memory");

        // ---- reload both P tiles as 16x32 A-fragments ----
        BF16Frag pf[2];
#pragma unroll
        for (int im = 0; im < 2; ++im) {
            pf[im].h[0] = *(const u16x8*)&ldsP[wave][im][r * 32 + half * 16];
            pf[im].h[1] = *(const u16x8*)&ldsP[wave][im][r * 32 + half * 16 + 8];
        }

        // ---- O += P * V ; denominator tile += P * ones (row sums) ----
#pragma unroll
        for (int t = 0; t < 4; ++t) {
            BF16Frag vf;
            vf.u = *(const u16x16*)(Vt + vbase + (size_t)(t * 16) * 2048 + j0);
            O[0][t] = wmma_bf16(pf[0].v, vf.v, O[0][t]);
            O[1][t] = wmma_bf16(pf[1].v, vf.v, O[1][t]);
        }
        O[0][4] = wmma_bf16(pf[0].v, ones.v, O[0][4]);
        O[1][4] = wmma_bf16(pf[1].v, ones.v, O[1][4]);
    }

    // ---- epilogue: normalize by the WMMA-accumulated row sums, store ----
#pragma unroll
    for (int im = 0; im < 2; ++im) {
        float inv[8];
#pragma unroll
        for (int v = 0; v < 8; ++v) inv[v] = 1.0f / O[im][4][v];
#pragma unroll
        for (int t = 0; t < 4; ++t)
#pragma unroll
            for (int v = 0; v < 8; ++v) {
                const int tok = q0 + im * 16 + v + half * 8;
                Ob[(size_t)(b * 2048 + tok) * 1024 + h * 64 + t * 16 + r] =
                    f2bf(O[im][t][v] * inv[v]);
            }
    }
}

// ---------------------------------------------------------------------------
// Host-side orchestration.
// Inputs: 0=query 1=key 2=value 3=mask 4=Wq 5=Wk 6=Wv 7=Wo
// ---------------------------------------------------------------------------
extern "C" void kernel_launch(void* const* d_in, const int* in_sizes, int n_in,
                              void* d_out, int out_size, void* d_ws, size_t ws_size,
                              hipStream_t stream) {
    (void)in_sizes; (void)n_in; (void)out_size; (void)ws_size;

    const float* query = (const float*)d_in[0];
    const float* key   = (const float*)d_in[1];
    const float* value = (const float*)d_in[2];
    const int*   mask  = (const int*)d_in[3];
    const float* Wq    = (const float*)d_in[4];
    const float* Wk    = (const float*)d_in[5];
    const float* Wv    = (const float*)d_in[6];
    const float* Wo    = (const float*)d_in[7];

    char* ws = (char*)d_ws;
    const size_t WT_BYTES = (size_t)1024 * 1024 * 2;         // 2 MB each
    const size_t XB_BYTES = (size_t)8192 * 1024 * 2;         // 16 MB each
    unsigned short* WqT = (unsigned short*)(ws + 0 * WT_BYTES);
    unsigned short* WkT = (unsigned short*)(ws + 1 * WT_BYTES);
    unsigned short* WvT = (unsigned short*)(ws + 2 * WT_BYTES);
    unsigned short* WoT = (unsigned short*)(ws + 3 * WT_BYTES);
    unsigned short* Qb  = (unsigned short*)(ws + 4 * WT_BYTES);
    unsigned short* Kb  = (unsigned short*)(ws + 4 * WT_BYTES + 1 * XB_BYTES);
    unsigned short* Vt  = (unsigned short*)(ws + 4 * WT_BYTES + 2 * XB_BYTES);
    unsigned short* Ab  = (unsigned short*)(ws + 4 * WT_BYTES + 3 * XB_BYTES);

    // 1) weight transpose+convert (f32 k-major -> bf16 n-major)
    dim3 tb(32, 8), tg(32, 32);
    wtrans_kernel<<<tg, tb, 0, stream>>>(Wq, WqT);
    wtrans_kernel<<<tg, tb, 0, stream>>>(Wk, WkT);
    wtrans_kernel<<<tg, tb, 0, stream>>>(Wv, WvT);
    wtrans_kernel<<<tg, tb, 0, stream>>>(Wo, WoT);

    // 2) projections: 8192x1024 @ 1024x1024, 64x64 tile per wave
    const int M = 8192, N = 1024, K = 1024;
    const int gemm_blocks = (M / 64) * (N / 64) / 8;         // 256
    gemm_wmma_kernel<1, 0><<<gemm_blocks, 256, 0, stream>>>(query, WqT, Qb, M, N, K);
    gemm_wmma_kernel<1, 0><<<gemm_blocks, 256, 0, stream>>>(key,   WkT, Kb, M, N, K);
    gemm_wmma_kernel<1, 1><<<gemm_blocks, 256, 0, stream>>>(value, WvT, Vt, M, N, K);

    // 3) flash attention: 4*16*64 = 4096 waves, 32 queries each
    attn_kernel<<<512, 256, 0, stream>>>(Qb, Kb, Vt, mask, Ab);

    // 4) output projection -> f32 d_out
    gemm_wmma_kernel<0, 2><<<gemm_blocks, 256, 0, stream>>>(Ab, WoT, d_out, M, N, K);
}
